// GraphMultiHeadAttention_38654705664487
// MI455X (gfx1250) — matrix operand order repaired and checked by reading, not compiled, not run on hardware
//
#include <hip/hip_runtime.h>
#include <hip/hip_bf16.h>

// Problem constants (from reference setup)
#define BGR   32          // graphs
#define NNODE 512         // nodes per graph
#define HIDD  512         // hidden dim
#define NH    8           // heads
#define HD    64          // head dim
#define EDIM  16          // edge feature dim
#define QB    64          // queries per attention block
#define NQB   (NNODE/QB)  // 8 query blocks per graph
#define NBUCKET (BGR*NQB) // 256 edge buckets

typedef __attribute__((ext_vector_type(16))) __bf16 v16bf;
typedef __attribute__((ext_vector_type(8)))  float  v8f;
typedef __attribute__((ext_vector_type(4)))  unsigned int u32x4;
typedef __attribute__((ext_vector_type(8)))  int i32x8;
typedef __attribute__((ext_vector_type(4)))  int i32x4;

#if defined(__has_builtin)
#if __has_builtin(__builtin_amdgcn_tensor_load_to_lds) && \
    __has_builtin(__builtin_amdgcn_s_wait_tensorcnt)
#define USE_TDM 1
#endif
#endif

// ---------------------------------------------------------------------------
// WMMA helpers (gfx1250 wave32 layouts per CDNA5 ISA 7.12.2)
// ---------------------------------------------------------------------------
static __device__ inline v8f wmma_bf16(v16bf a, v16bf b, v8f c) {
  return __builtin_amdgcn_wmma_f32_16x16x32_bf16(
      /*neg_a=*/false, a, /*neg_b=*/false, b,
      /*c_mod=*/(short)0, c, /*reuse_a=*/false, /*reuse_b=*/false);
}

// A fragment 16x32 (MxK) from row-major LDS tile. base = &tile[m0*ld + k0].
// lane 0-15: M=lane, K base 0; lane 16-31: M=lane-16, K base +8.
// element e -> K = kbase + e + (e>=8 ? 8 : 0) (pairs are memory-contiguous)
static __device__ inline v16bf frag_a_lds(const __bf16* base, int ld) {
  const int lane = threadIdx.x & 31;
  const __bf16* p = base + (lane & 15) * ld + ((lane >> 4) << 3);
  v16bf a;
#pragma unroll
  for (int e = 0; e < 16; e += 2) {
    int kk = e + ((e >> 3) << 3);
    a[e]     = p[kk];
    a[e + 1] = p[kk + 1];
  }
  return a;
}

// B fragment 32x16 (KxN) from LDS tile stored TRANSPOSED as Bt[n][k].
// base = &Bt[n0*ld + k0]. lane n = lane&15; K base = (lane>=16)?8:0; e -> K=kbase + e + (e>=8 ? 8 : 0).
static __device__ inline v16bf frag_b_lds(const __bf16* base, int ld) {
  const int lane = threadIdx.x & 31;
  const __bf16* p = base + (lane & 15) * ld + ((lane >> 4) << 3);
  v16bf b;
#pragma unroll
  for (int e = 0; e < 16; ++e) b[e] = p[e + ((e >> 3) << 3)];
  return b;
}

#ifdef USE_TDM
// ---------------------------------------------------------------------------
// Tensor Data Mover: DMA one 2-D bf16 tile (tile_d0 x tile_d1, row stride
// row_stride elems) from global into LDS, with optional LDS row padding.
// D# packing per CDNA5 ISA 8.3/8.4. Flat LDS addresses carry the LDS byte
// offset in bits [31:0] (ISA 10.2 aperture rules), so truncation is the
// lds_addr field. Issue from one wave only; EXEC is ignored by TDM.
// ---------------------------------------------------------------------------
static __device__ inline void tdm_load_2d_bf16(const void* lds_dst,
                                               const void* gsrc,
                                               unsigned tile_d0, unsigned tile_d1,
                                               unsigned tensor_d0, unsigned tensor_d1,
                                               unsigned long long row_stride,
                                               unsigned pad_interval,  // 1<<v 8B units
                                               unsigned pad_amount,    // (v+1) dwords
                                               bool pad_en) {
  unsigned long long ga = (unsigned long long)(uintptr_t)gsrc;
  unsigned lds_off = (unsigned)(uintptr_t)lds_dst;
  u32x4 g0;
  g0[0] = 1u;                                  // count=1, user descriptor
  g0[1] = lds_off;                             // lds_addr (bytes)
  g0[2] = (unsigned)ga;                        // global_addr[31:0]
  g0[3] = (unsigned)((ga >> 32) & 0x1FFFFFFull) | (2u << 30);  // [56:32] | type=2
  i32x8 g1;
  unsigned w0 = (1u << 16);                    // data_size=1 (2 bytes/elem)
  if (pad_en)
    w0 |= (1u << 20) | (pad_interval << 22) | (pad_amount << 25);
  g1[0] = (int)w0;
  g1[1] = (int)((tensor_d0 & 0xFFFFu) << 16);  // atomic_addr=0 | tdim0 lo16
  g1[2] = (int)(((tensor_d0 >> 16) & 0xFFFFu) | ((tensor_d1 & 0xFFFFu) << 16));
  g1[3] = (int)(((tensor_d1 >> 16) & 0xFFFFu) | ((tile_d0 & 0xFFFFu) << 16));
  g1[4] = (int)(tile_d1 & 0xFFFFu);            // tile_dim1 | tile_dim2=0
  g1[5] = (int)(unsigned)(row_stride & 0xFFFFFFFFull);          // d0 stride lo32
  g1[6] = (int)(unsigned)((row_stride >> 32) & 0xFFFFull);      // d0 stride hi16
  g1[7] = 0;
  i32x4 g2 = {};                               // dims 2/3 unused (tile_dim3=0)
  i32x4 g3 = {};
#if __clang_major__ >= 23
  i32x8 gz = {};
  __builtin_amdgcn_tensor_load_to_lds(g0, g1, g2, g3, gz, 0);
#else
  __builtin_amdgcn_tensor_load_to_lds(g0, g1, g2, g3, 0);
#endif
}
#endif  // USE_TDM

// ---------------------------------------------------------------------------
// Elementwise f32 -> bf16
// ---------------------------------------------------------------------------
__global__ void f32_to_bf16_kernel(const float* __restrict__ in,
                                   __bf16* __restrict__ out, size_t n) {
  size_t i = (size_t)blockIdx.x * blockDim.x + threadIdx.x;
  if (i < n) out[i] = (__bf16)in[i];
}

// ---------------------------------------------------------------------------
// Per-edge head bias: eb[e][h] = edge_attr[e] . We[:,h] + be[h]
// ---------------------------------------------------------------------------
__global__ void edge_bias_kernel(const float* __restrict__ ea,
                                 const float* __restrict__ We,
                                 const float* __restrict__ be,
                                 float* __restrict__ eb, int E) {
  int e = blockIdx.x * blockDim.x + threadIdx.x;
  if (e >= E) return;
  float f[EDIM];
#pragma unroll
  for (int j = 0; j < EDIM; ++j) f[j] = ea[(size_t)e * EDIM + j];
#pragma unroll
  for (int h = 0; h < NH; ++h) {
    float s = be[h];
#pragma unroll
    for (int j = 0; j < EDIM; ++j) s += f[j] * We[j * NH + h];
    eb[(size_t)e * NH + h] = s;
  }
}

// ---------------------------------------------------------------------------
// Edge bucketing: bucket = graph(src) * NQB + (loc_src / QB)
// ---------------------------------------------------------------------------
static __device__ inline int edge_bucket(int src) {
  return (src >> 9) * NQB + ((src & (NNODE - 1)) >> 6);
}

__global__ void edge_count_kernel(const int* __restrict__ ei,
                                  int* __restrict__ counts, int E) {
  int e = blockIdx.x * blockDim.x + threadIdx.x;
  if (e >= E) return;
  atomicAdd(&counts[edge_bucket(ei[e])], 1);
}

__global__ void scan_kernel(const int* __restrict__ counts,
                            int* __restrict__ offsets) {
  __shared__ int tmp[NBUCKET];
  int t = threadIdx.x;
  tmp[t] = counts[t];
  __syncthreads();
  for (int d = 1; d < NBUCKET; d <<= 1) {
    int vv = (t >= d) ? tmp[t - d] : 0;
    __syncthreads();
    tmp[t] += vv;
    __syncthreads();
  }
  offsets[t + 1] = tmp[t];
  if (t == 0) offsets[0] = 0;
}

__global__ void edge_fill_kernel(const int* __restrict__ ei,
                                 const int* __restrict__ offsets,
                                 int* __restrict__ cursor,
                                 int* __restrict__ ids, int E) {
  int e = blockIdx.x * blockDim.x + threadIdx.x;
  if (e >= E) return;
  int bkt = edge_bucket(ei[e]);
  int pos = atomicAdd(&cursor[bkt], 1);
  ids[offsets[bkt] + pos] = e;
}

// ---------------------------------------------------------------------------
// Tiled bf16 WMMA GEMM: C[M][N] = A[M][K] * W[K][N] + bias
// block tile 128x128, 8 waves, wave tile 64x32 (4x2 WMMA tiles).
// A tile staged by TDM (row pad 64B -> +8B gives ld 36); W tile transposed
// through VGPRs (TDM cannot transpose).
// ---------------------------------------------------------------------------
template <bool OUT_F32>
__global__ __launch_bounds__(256)
void gemm_bf16_kernel(const __bf16* __restrict__ A, const __bf16* __restrict__ W,
                      const float* __restrict__ bias, void* __restrict__ Cout,
                      int M, int K, int Ncol) {
  __shared__ __bf16 As[128][36];   // row-major A tile, 2-dword row pad
  __shared__ __bf16 Wt[128][40];   // transposed W tile: Wt[n][k], padded

  const int tid  = threadIdx.x;
  const int wave = tid >> 5;
  const int lane = tid & 31;
  const int m_blk = blockIdx.y * 128;
  const int n_blk = blockIdx.x * 128;
  const int wr = wave >> 2;             // 0..1
  const int wc = wave & 3;              // 0..3
  const int wm = wr * 64;               // wave M base within block
  const int wn = wc * 32;               // wave N base within block

  v8f acc[4][2];
#pragma unroll
  for (int i = 0; i < 4; ++i)
#pragma unroll
    for (int j = 0; j < 2; ++j) acc[i][j] = (v8f){};

  for (int kk = 0; kk < K; kk += 32) {
#ifdef USE_TDM
    // A tile 128x32: single TDM descriptor; pad_interval=3 (64B rows),
    // pad_amount=1 (+2 dwords) -> LDS leading dim 36 elements.
    if (wave == 0) {
      tdm_load_2d_bf16(&As[0][0], A + (size_t)m_blk * K + kk,
                       /*tile_d0=*/32, /*tile_d1=*/128,
                       /*tensor_d0=*/(unsigned)K, /*tensor_d1=*/(unsigned)M,
                       /*row_stride=*/(unsigned long long)K,
                       /*pad_interval=*/3, /*pad_amount=*/1, /*pad_en=*/true);
    }
#else
    {
      int row = tid >> 1, col = (tid & 1) << 4;
      const __bf16* g = A + (size_t)(m_blk + row) * K + kk + col;
#pragma unroll
      for (int j = 0; j < 16; ++j) As[row][col + j] = g[j];
      if (kk + 32 < K) __builtin_prefetch(g + 32, 0, 1);  // global_prefetch_b8
    }
#endif
    // W tile 32x128 -> transposed into Wt[n][k]
    {
      int k = tid >> 3, ng = (tid & 7) << 4;
      const __bf16* g = W + (size_t)(kk + k) * Ncol + n_blk + ng;
#pragma unroll
      for (int j = 0; j < 16; ++j) Wt[ng + j][k] = g[j];
    }
#ifdef USE_TDM
    if (wave == 0) __builtin_amdgcn_s_wait_tensorcnt(0);
#endif
    __syncthreads();

    v16bf bfr[2];
#pragma unroll
    for (int j = 0; j < 2; ++j)
      bfr[j] = frag_b_lds(&Wt[wn + j * 16][0], 40);
#pragma unroll
    for (int i = 0; i < 4; ++i) {
      v16bf afr = frag_a_lds(&As[wm + i * 16][0], 36);
#pragma unroll
      for (int j = 0; j < 2; ++j)
        acc[i][j] = wmma_bf16(afr, bfr[j], acc[i][j]);
    }
    __syncthreads();
  }

  // store (C/D layout: M = r + 8*(lane>=16), N = lane&15)
  const int cn = lane & 15;
  const int rb = (lane >> 4) << 3;
#pragma unroll
  for (int i = 0; i < 4; ++i) {
#pragma unroll
    for (int j = 0; j < 2; ++j) {
      int col = n_blk + wn + j * 16 + cn;
      float bvv = bias[col];
#pragma unroll
      for (int r = 0; r < 8; ++r) {
        int row = m_blk + wm + i * 16 + rb + r;
        float val = acc[i][j][r] + bvv;
        if (OUT_F32)
          ((float*)Cout)[(size_t)row * Ncol + col] = val;
        else
          ((__bf16*)Cout)[(size_t)row * Ncol + col] = (__bf16)val;
      }
    }
  }
}

// ---------------------------------------------------------------------------
// Attention: one block per (graph b, head h, 64-query block qb)
// S = scale*Q K^T (+edge bias) -> softmax -> O = P V   (all WMMA)
// Q and K tiles staged by TDM (row pad 128B -> +32B gives ld 80).
// ---------------------------------------------------------------------------
#define LDT   80   // bf16 tile leading dim (64 + 16 pad)
#define LDSS  520  // f32 score leading dim (512 + 8 pad)
#define LDP   528  // bf16 P leading dim (512 + 16 pad)
#define ATTN_SMEM_BYTES (3 * 64 * LDT * 2 + 64 * LDSS * 4 + 64 * LDP * 2)

__global__ __launch_bounds__(256)
void attn_kernel(const __bf16* __restrict__ q, const __bf16* __restrict__ k,
                 const __bf16* __restrict__ vv, const float* __restrict__ eb,
                 const int* __restrict__ ei, const int* __restrict__ offsets,
                 const int* __restrict__ ids, __bf16* __restrict__ outw, int E) {
  extern __shared__ char smem[];
  __bf16* Qs = (__bf16*)smem;            // [64][LDT]  query tile
  __bf16* Ks = Qs + 64 * LDT;            // [64][LDT]  key tile (natural [key][d])
  __bf16* Vt = Ks + 64 * LDT;            // [64][LDT]  value tile transposed [d][key]
  float*  S  = (float*)(Vt + 64 * LDT);  // [64][LDSS] scores f32
  __bf16* P  = (__bf16*)(S + 64 * LDSS); // [64][LDP]  softmax probs bf16

  const int tid  = threadIdx.x;
  const int wave = tid >> 5;
  const int lane = tid & 31;
  const int blk = blockIdx.x;
  const int qb = blk & (NQB - 1);
  const int h  = (blk >> 3) & (NH - 1);
  const int b  = blk >> 6;
  const int node0 = b * NNODE + qb * QB;  // first query node (global row)
  const int key0g = b * NNODE;            // first key node of graph

  // load Q tile [64 queries x 64 d]
#ifdef USE_TDM
  if (wave == 0) {
    tdm_load_2d_bf16(Qs, q + (size_t)node0 * HIDD + h * HD,
                     /*tile_d0=*/64, /*tile_d1=*/64,
                     /*tensor_d0=*/HIDD, /*tensor_d1=*/BGR * NNODE,
                     /*row_stride=*/HIDD,
                     /*pad_interval=*/4, /*pad_amount=*/7, /*pad_en=*/true);
    __builtin_amdgcn_s_wait_tensorcnt(0);
  }
#else
  {
    int row = tid >> 2, cg = (tid & 3) << 4;
    const __bf16* g = q + (size_t)(node0 + row) * HIDD + h * HD + cg;
#pragma unroll
    for (int j = 0; j < 16; ++j) Qs[row * LDT + cg + j] = g[j];
  }
#endif
  __syncthreads();

  // ---- S = scale * Q K^T over all 512 keys, 64-key tiles ----
  const float scale = 0.125f;  // 1/sqrt(64)
  const int mt  = wave & 3;    // query 16-tile 0..3
  const int ntp = wave >> 2;   // pair selector for key/d tiles

  for (int kt = 0; kt < NNODE; kt += 64) {
#ifdef USE_TDM
    if (wave == 0) {
      tdm_load_2d_bf16(Ks, k + (size_t)(key0g + kt) * HIDD + h * HD,
                       64, 64, HIDD, BGR * NNODE, HIDD, 4, 7, true);
      __builtin_amdgcn_s_wait_tensorcnt(0);
    }
#else
    {
      int row = tid >> 2, cg = (tid & 3) << 4;
      const __bf16* g = k + (size_t)(key0g + kt + row) * HIDD + h * HD + cg;
#pragma unroll
      for (int j = 0; j < 16; ++j) Ks[row * LDT + cg + j] = g[j];
    }
#endif
    __syncthreads();
#pragma unroll
    for (int jt = 0; jt < 2; ++jt) {
      int nt = (ntp << 1) + jt;
      v8f acc = (v8f){};
#pragma unroll
      for (int ks = 0; ks < 64; ks += 32) {
        v16bf af = frag_a_lds(&Qs[mt * 16 * LDT + ks], LDT);
        // Ks is [key][d]: exactly Bt[n=key][k=d] for S[m][key]
        v16bf bf = frag_b_lds(&Ks[nt * 16 * LDT + ks], LDT);
        acc = wmma_bf16(af, bf, acc);
      }
      int cn = lane & 15, rb = (lane >> 4) << 3;
#pragma unroll
      for (int r = 0; r < 8; ++r)
        S[(mt * 16 + rb + r) * LDSS + kt + nt * 16 + cn] = acc[r] * scale;
    }
    __syncthreads();
  }

  // ---- edge-bias scatter-add into LDS scores ----
  {
    int bkt = b * NQB + qb;
    int beg = offsets[bkt], end = offsets[bkt + 1];
    for (int idx = beg + tid; idx < end; idx += 256) {
      int e   = ids[idx];
      int src = ei[e];
      int dst = ei[E + e];
      int lr  = (src & (NNODE - 1)) - qb * QB;  // 0..63 by bucketing
      int lc  = dst & (NNODE - 1);
      atomicAdd(&S[lr * LDSS + lc], eb[(size_t)e * NH + h]);
    }
  }
  __syncthreads();

  // ---- softmax over keys; each wave owns 8 rows; write P as bf16 ----
  for (int ri = 0; ri < 8; ++ri) {
    int row = wave * 8 + ri;
    float* sr = &S[row * LDSS];
    float mx = -3.0e38f;
#pragma unroll
    for (int j = 0; j < 16; ++j) mx = fmaxf(mx, sr[lane + 32 * j]);
#pragma unroll
    for (int off = 16; off > 0; off >>= 1) mx = fmaxf(mx, __shfl_xor(mx, off, 32));
    float vals[16];
    float sum = 0.0f;
#pragma unroll
    for (int j = 0; j < 16; ++j) {
      vals[j] = __expf(sr[lane + 32 * j] - mx);
      sum += vals[j];
    }
#pragma unroll
    for (int off = 16; off > 0; off >>= 1) sum += __shfl_xor(sum, off, 32);
    float inv = 1.0f / sum;
    __bf16* pr = &P[row * LDP];
#pragma unroll
    for (int j = 0; j < 16; ++j) pr[lane + 32 * j] = (__bf16)(vals[j] * inv);
  }
  __syncthreads();

  // ---- O = P V : wave owns (mt, d-tiles {2*ntp, 2*ntp+1}) ----
  v8f oacc[2];
  oacc[0] = (v8f){};
  oacc[1] = (v8f){};
  for (int kt = 0; kt < NNODE; kt += 64) {
    // V tile transposed into Vt[d][key]; must go through VGPRs
    {
      int key = tid >> 2, dg = (tid & 3) << 4;
      const __bf16* g = vv + (size_t)(key0g + kt + key) * HIDD + h * HD + dg;
#pragma unroll
      for (int j = 0; j < 16; ++j) Vt[(dg + j) * LDT + key] = g[j];
    }
    __syncthreads();
#pragma unroll
    for (int ks = 0; ks < 64; ks += 32) {
      v16bf af = frag_a_lds(&P[mt * 16 * LDP + kt + ks], LDP);
#pragma unroll
      for (int jt = 0; jt < 2; ++jt) {
        int dt = (ntp << 1) + jt;
        v16bf bf = frag_b_lds(&Vt[dt * 16 * LDT + ks], LDT);
        oacc[jt] = wmma_bf16(af, bf, oacc[jt]);
      }
    }
    __syncthreads();
  }
  // store O tile to attn_out (bf16); (b,qb) rows x h cols are disjoint
  {
    int cn = lane & 15, rb = (lane >> 4) << 3;
#pragma unroll
    for (int jt = 0; jt < 2; ++jt) {
      int dt = (ntp << 1) + jt;
#pragma unroll
      for (int r = 0; r < 8; ++r) {
        int row = node0 + mt * 16 + rb + r;
        int col = h * HD + dt * 16 + cn;
        outw[(size_t)row * HIDD + col] = (__bf16)oacc[jt][r];
      }
    }
  }
}

// ---------------------------------------------------------------------------
// Host launcher
// ---------------------------------------------------------------------------
extern "C" void kernel_launch(void* const* d_in, const int* in_sizes, int n_in,
                              void* d_out, int out_size, void* d_ws, size_t ws_size,
                              hipStream_t stream) {
  (void)n_in; (void)out_size; (void)ws_size;
  const float* x  = (const float*)d_in[0];
  const int*   ei = (const int*)d_in[2];     // [2][E]: src row, dst row
  const float* ea = (const float*)d_in[3];
  const float* Wq = (const float*)d_in[4];
  const float* bq = (const float*)d_in[5];
  const float* Wk = (const float*)d_in[6];
  const float* bk = (const float*)d_in[7];
  const float* Wv = (const float*)d_in[8];
  const float* bv = (const float*)d_in[9];
  const float* Wo = (const float*)d_in[10];
  const float* bo = (const float*)d_in[11];
  const float* We = (const float*)d_in[12];
  const float* be = (const float*)d_in[13];
  const int E = in_sizes[2] / 2;
  const int M = BGR * NNODE;  // 16384 rows

  // workspace carve (256B aligned)
  char* wsp = (char*)d_ws;
  auto take = [&](size_t bytes) {
    char* p = wsp;
    wsp += (bytes + 255) & ~(size_t)255;
    return p;
  };
  __bf16* xb  = (__bf16*)take((size_t)M * HIDD * 2);
  __bf16* wqb = (__bf16*)take((size_t)HIDD * HIDD * 2);
  __bf16* wkb = (__bf16*)take((size_t)HIDD * HIDD * 2);
  __bf16* wvb = (__bf16*)take((size_t)HIDD * HIDD * 2);
  __bf16* wob = (__bf16*)take((size_t)HIDD * HIDD * 2);
  __bf16* qw  = (__bf16*)take((size_t)M * HIDD * 2);
  __bf16* kw  = (__bf16*)take((size_t)M * HIDD * 2);
  __bf16* vw  = (__bf16*)take((size_t)M * HIDD * 2);
  __bf16* ao  = (__bf16*)take((size_t)M * HIDD * 2);
  float*  eb  = (float*)take((size_t)E * NH * 4);
  int* counts  = (int*)take(NBUCKET * 4);
  int* cursor  = (int*)take(NBUCKET * 4);
  int* offsets = (int*)take((NBUCKET + 1) * 4);
  int* ids     = (int*)take((size_t)E * 4);

  // 1) precision conversions
  {
    size_t nx = (size_t)M * HIDD;
    f32_to_bf16_kernel<<<(nx + 255) / 256, 256, 0, stream>>>(x, xb, nx);
    size_t nw = (size_t)HIDD * HIDD;
    f32_to_bf16_kernel<<<(nw + 255) / 256, 256, 0, stream>>>(Wq, wqb, nw);
    f32_to_bf16_kernel<<<(nw + 255) / 256, 256, 0, stream>>>(Wk, wkb, nw);
    f32_to_bf16_kernel<<<(nw + 255) / 256, 256, 0, stream>>>(Wv, wvb, nw);
    f32_to_bf16_kernel<<<(nw + 255) / 256, 256, 0, stream>>>(Wo, wob, nw);
  }

  // 2) edge biases + buckets
  edge_bias_kernel<<<(E + 255) / 256, 256, 0, stream>>>(ea, We, be, eb, E);
  hipMemsetAsync(counts, 0, NBUCKET * 4, stream);
  hipMemsetAsync(cursor, 0, NBUCKET * 4, stream);
  edge_count_kernel<<<(E + 255) / 256, 256, 0, stream>>>(ei, counts, E);
  scan_kernel<<<1, NBUCKET, 0, stream>>>(counts, offsets);
  edge_fill_kernel<<<(E + 255) / 256, 256, 0, stream>>>(ei, offsets, cursor, ids, E);

  // 3) Q/K/V projections (WMMA bf16, f32 accumulate, bf16 out)
  dim3 gg(HIDD / 128, M / 128);
  gemm_bf16_kernel<false><<<gg, 256, 0, stream>>>(xb, wqb, bq, qw, M, HIDD, HIDD);
  gemm_bf16_kernel<false><<<gg, 256, 0, stream>>>(xb, wkb, bk, kw, M, HIDD, HIDD);
  gemm_bf16_kernel<false><<<gg, 256, 0, stream>>>(xb, wvb, bv, vw, M, HIDD, HIDD);

  // 4) attention (one block per (graph, head, 64-query block))
  hipFuncSetAttribute(reinterpret_cast<const void*>(attn_kernel),
                      hipFuncAttributeMaxDynamicSharedMemorySize, ATTN_SMEM_BYTES);
  attn_kernel<<<BGR * NH * NQB, 256, ATTN_SMEM_BYTES, stream>>>(
      qw, kw, vw, eb, ei, offsets, ids, ao, E);

  // 5) output projection -> f32 d_out
  gemm_bf16_kernel<true><<<gg, 256, 0, stream>>>(ao, wob, bo, (float*)d_out,
                                                 M, HIDD, HIDD);
}
